// Attention_17798344474999
// MI455X (gfx1250) — matrix operand order repaired and checked by reading, not compiled, not run on hardware
//
#include <hip/hip_runtime.h>

// ---------------- types ----------------
typedef __bf16 v16bf __attribute__((ext_vector_type(16)));
typedef __bf16 v8bf  __attribute__((ext_vector_type(8)));
typedef __bf16 v4bf  __attribute__((ext_vector_type(4)));
typedef float  v8f   __attribute__((ext_vector_type(8)));

#define DIM   1024
#define LQ    2048
#define NH    16
#define HD    64
#define BATCH 2
#define SCALE 0.125f   // 64^-0.5

union V16U { v16bf v; v8bf h[2]; };

static __device__ __forceinline__ v8f wmma_bf16(v16bf a, v16bf b, v8f c) {
  return __builtin_amdgcn_wmma_f32_16x16x32_bf16(false, a, false, b, (short)0, c,
                                                 false, false);
}

// A fragment: 16x32 bf16 tile at (rows base .. base+15*ld, cols k0..k0+31).
// Layout: lane m(0-15) holds K = 0..15, lane m+16 holds K = 16..31 (matches loadB).
static __device__ __forceinline__ v16bf loadA(const __bf16* base, int ld, int k0, int lane) {
  int m  = lane & 15;
  int hi = lane >> 4;
  const __bf16* p = base + (size_t)m * ld + k0 + hi * 16;
  V16U u;
  u.h[0] = *(const v8bf*)(p);
  u.h[1] = *(const v8bf*)(p + 8);
  return u.v;
}

// B fragment: 32(K) x 16(N); B[k][n] = W[n][k] with W row-major [N, K].
// Lanes 0-15 hold K=0..15 for N=lane, lanes 16-31 hold K=16..31.
static __device__ __forceinline__ v16bf loadB(const __bf16* base, int ld, int k0, int lane) {
  int n    = lane & 15;
  int koff = (lane >> 4) * 16;
  const __bf16* p = base + (size_t)n * ld + k0 + koff;
  return *(const v16bf*)p;
}

// Ping-pong pipelined 32x64 GEMM mainloop: load one buffer while 8 WMMAs
// consume the other. Base pointers are wave-uniform (SGPR) -> saddr loads.
static __device__ __forceinline__ void gemm_mainloop(
    const __bf16* A0, const __bf16* A1, const __bf16* const (&Wn)[4],
    int lane, v8f (&acc)[2][4]) {
  v16bf aX0, aX1, bX[4], aY0, aY1, bY[4];
  aX0 = loadA(A0, DIM, 0, lane);
  aX1 = loadA(A1, DIM, 0, lane);
  #pragma unroll
  for (int t = 0; t < 4; ++t) bX[t] = loadB(Wn[t], DIM, 0, lane);

  for (int k0 = 0; k0 < DIM; k0 += 64) {
    // stage A: load Y at k0+32, compute with X
    {
      int kn = k0 + 32;
      aY0 = loadA(A0, DIM, kn, lane);
      aY1 = loadA(A1, DIM, kn, lane);
      #pragma unroll
      for (int t = 0; t < 4; ++t) bY[t] = loadB(Wn[t], DIM, kn, lane);
      #pragma unroll
      for (int t = 0; t < 4; ++t) acc[0][t] = wmma_bf16(aX0, bX[t], acc[0][t]);
      #pragma unroll
      for (int t = 0; t < 4; ++t) acc[1][t] = wmma_bf16(aX1, bX[t], acc[1][t]);
    }
    // stage B: load X at k0+64 (wraps harmlessly on last iter), compute with Y
    {
      int kn = (k0 + 64) & (DIM - 1);
      aX0 = loadA(A0, DIM, kn, lane);
      aX1 = loadA(A1, DIM, kn, lane);
      #pragma unroll
      for (int t = 0; t < 4; ++t) bX[t] = loadB(Wn[t], DIM, kn, lane);
      #pragma unroll
      for (int t = 0; t < 4; ++t) acc[0][t] = wmma_bf16(aY0, bY[t], acc[0][t]);
      #pragma unroll
      for (int t = 0; t < 4; ++t) acc[1][t] = wmma_bf16(aY1, bY[t], acc[1][t]);
    }
  }
}

// ---------------- fp32 -> bf16 convert ----------------
__global__ void cvt_bf16(const float* __restrict__ s, __bf16* __restrict__ d, int n4) {
  int i = blockIdx.x * blockDim.x + threadIdx.x;
  if (i < n4) {
    float4 f = ((const float4*)s)[i];
    v4bf o = { (__bf16)f.x, (__bf16)f.y, (__bf16)f.z, (__bf16)f.w };
    ((v4bf*)d)[i] = o;
  }
}

// ---------------- QKV projection: [4096,1024] x [3072,1024]^T ----------------
// Writes Q (scaled) and K as [B*H][L][D], V transposed as [B*H][D][L], all bf16.
__global__ void __launch_bounds__(256)
qkv_gemm(const __bf16* __restrict__ X, const __bf16* __restrict__ W,
         __bf16* __restrict__ Q, __bf16* __restrict__ Kk, __bf16* __restrict__ Vt) {
  int lane = threadIdx.x & 31;
  // readfirstlane -> wave id (and everything derived) is provably wave-uniform,
  // so base pointers live in SGPRs and loads use the saddr form.
  int wave = __builtin_amdgcn_readfirstlane(threadIdx.x >> 5);
  int m0 = blockIdx.y * 64  + (wave >> 2) * 32;
  int n0 = blockIdx.x * 256 + (wave & 3) * 64;

  const __bf16* A0 = X + (size_t)m0 * DIM;
  const __bf16* A1 = X + (size_t)(m0 + 16) * DIM;
  const __bf16* Wn[4];
  #pragma unroll
  for (int t = 0; t < 4; ++t) Wn[t] = W + (size_t)(n0 + t * 16) * DIM;

  v8f acc[2][4] = {};
  gemm_mainloop(A0, A1, Wn, lane, acc);

  // n0 is 64-aligned -> whole wave tile is one (s, h); s,h are scalar
  int s = n0 >> 10;                 // 0=q 1=k 2=v
  int h = (n0 >> 6) & 15;
  int nloc  = lane & 15;
  int basem = (lane >> 4) * 8;
  #pragma unroll
  for (int mi = 0; mi < 2; ++mi) {
    #pragma unroll
    for (int t = 0; t < 4; ++t) {
      int d = t * 16 + nloc;
      #pragma unroll
      for (int r = 0; r < 8; ++r) {
        int m = m0 + mi * 16 + basem + r;
        int b = m >> 11, l = m & (LQ - 1);
        int bh = b * NH + h;
        float c = acc[mi][t][r];
        if (s == 0)      Q [((size_t)bh * LQ + l) * HD + d] = (__bf16)(c * SCALE);
        else if (s == 1) Kk[((size_t)bh * LQ + l) * HD + d] = (__bf16)c;
        else             Vt[((size_t)bh * HD + d) * LQ + l] = (__bf16)c;
      }
    }
  }
}

// ---------------- Flash attention: per (b,h), 64 query rows / block ----------------
__global__ void __launch_bounds__(128)
attn_kernel(const __bf16* __restrict__ Q, const __bf16* __restrict__ Kk,
            const __bf16* __restrict__ Vt, __bf16* __restrict__ O) {
  __shared__ __align__(32) __bf16 pshm[4][16][32];   // P tile staging per wave
  int lane = threadIdx.x & 31;
  int wave = __builtin_amdgcn_readfirstlane(threadIdx.x >> 5);  // 4 waves, 16 q-rows each
  int bh = blockIdx.z * NH + blockIdx.y;
  int q0 = blockIdx.x * 64 + wave * 16;

  const __bf16* Qb = Q  + (size_t)bh * LQ * HD;
  const __bf16* Kb = Kk + (size_t)bh * LQ * HD;
  const __bf16* Vb = Vt + (size_t)bh * HD * LQ;
  __bf16 (*myp)[32] = pshm[wave];

  v16bf qa0 = loadA(Qb + (size_t)q0 * HD, HD, 0,  lane);
  v16bf qa1 = loadA(Qb + (size_t)q0 * HD, HD, 32, lane);

  float mrow[8], lrow[8];
  #pragma unroll
  for (int r = 0; r < 8; ++r) { mrow[r] = -1e30f; lrow[r] = 0.0f; }
  v8f acc[4] = {};
  int nloc  = lane & 15;
  int basem = (lane >> 4) * 8;

  // One 32-key block: S via WMMA from current K frags, online softmax,
  // P relayout through LDS, P*V via WMMA. Loads next K frags into (kn0,kn1).
  auto stage = [&](int j, int jn,
                   v16bf (&kc0)[2], v16bf (&kc1)[2],
                   v16bf (&kn0)[2], v16bf (&kn1)[2]) {
    v16bf vb[4];
    #pragma unroll
    for (int t = 0; t < 4; ++t) vb[t] = loadB(Vb + (size_t)(t * 16) * LQ, LQ, j, lane);
    kn0[0] = loadB(Kb + (size_t)jn * HD,        HD, 0,  lane);
    kn0[1] = loadB(Kb + (size_t)jn * HD,        HD, 32, lane);
    kn1[0] = loadB(Kb + (size_t)(jn + 16) * HD, HD, 0,  lane);
    kn1[1] = loadB(Kb + (size_t)(jn + 16) * HD, HD, 32, lane);

    v8f z = {};
    v8f s0 = wmma_bf16(qa0, kc0[0], z);
    s0     = wmma_bf16(qa1, kc0[1], s0);
    v8f s1 = wmma_bf16(qa0, kc1[0], z);
    s1     = wmma_bf16(qa1, kc1[1], s1);

    #pragma unroll
    for (int r = 0; r < 8; ++r) {
      float t = fmaxf(s0[r], s1[r]);
      t = fmaxf(t, __shfl_xor(t, 1, 32));
      t = fmaxf(t, __shfl_xor(t, 2, 32));
      t = fmaxf(t, __shfl_xor(t, 4, 32));
      t = fmaxf(t, __shfl_xor(t, 8, 32));
      float mn = fmaxf(mrow[r], t);
      float alpha = __expf(mrow[r] - mn);
      mrow[r] = mn;
      float p0 = __expf(s0[r] - mn);
      float p1 = __expf(s1[r] - mn);
      float ps = p0 + p1;
      ps += __shfl_xor(ps, 1, 32);
      ps += __shfl_xor(ps, 2, 32);
      ps += __shfl_xor(ps, 4, 32);
      ps += __shfl_xor(ps, 8, 32);
      lrow[r] = lrow[r] * alpha + ps;
      acc[0][r] *= alpha; acc[1][r] *= alpha;
      acc[2][r] *= alpha; acc[3][r] *= alpha;
      myp[basem + r][nloc]      = (__bf16)p0;
      myp[basem + r][16 + nloc] = (__bf16)p1;
    }

    v16bf pa = loadA(&myp[0][0], 32, 0, lane);
    #pragma unroll
    for (int t = 0; t < 4; ++t) acc[t] = wmma_bf16(pa, vb[t], acc[t]);
  };

  // prime buffer X with keys 0..31
  v16bf kX0[2], kX1[2], kY0[2], kY1[2];
  kX0[0] = loadB(Kb,           HD, 0,  lane);
  kX0[1] = loadB(Kb,           HD, 32, lane);
  kX1[0] = loadB(Kb + 16 * HD, HD, 0,  lane);
  kX1[1] = loadB(Kb + 16 * HD, HD, 32, lane);

  for (int j = 0; j < LQ; j += 64) {
    stage(j,      j + 32,              kX0, kX1, kY0, kY1);
    stage(j + 32, (j + 64) & (LQ - 1), kY0, kY1, kX0, kX1);
  }

  // normalize and write O as bf16 [B, L, H*D]
  int b = bh >> 4, h = bh & 15;
  #pragma unroll
  for (int r = 0; r < 8; ++r) {
    float inv = 1.0f / lrow[r];
    int l = q0 + basem + r;
    size_t rowoff = ((size_t)b * LQ + l) * DIM + h * HD;
    #pragma unroll
    for (int t = 0; t < 4; ++t)
      O[rowoff + t * 16 + nloc] = (__bf16)(acc[t][r] * inv);
  }
}

// ---------------- output projection: [4096,1024] x [1024,1024]^T + bias ----------------
__global__ void __launch_bounds__(256)
proj_gemm(const __bf16* __restrict__ A, const __bf16* __restrict__ W,
          const float* __restrict__ bias, float* __restrict__ out) {
  int lane = threadIdx.x & 31;
  int wave = __builtin_amdgcn_readfirstlane(threadIdx.x >> 5);
  int m0 = blockIdx.y * 64  + (wave >> 2) * 32;
  int n0 = blockIdx.x * 256 + (wave & 3) * 64;

  const __bf16* A0 = A + (size_t)m0 * DIM;
  const __bf16* A1 = A + (size_t)(m0 + 16) * DIM;
  const __bf16* Wn[4];
  #pragma unroll
  for (int t = 0; t < 4; ++t) Wn[t] = W + (size_t)(n0 + t * 16) * DIM;

  v8f acc[2][4] = {};
  gemm_mainloop(A0, A1, Wn, lane, acc);

  int nloc  = lane & 15;
  int basem = (lane >> 4) * 8;
  #pragma unroll
  for (int mi = 0; mi < 2; ++mi) {
    #pragma unroll
    for (int t = 0; t < 4; ++t) {
      int n = n0 + t * 16 + nloc;
      float bv = bias[n];
      #pragma unroll
      for (int r = 0; r < 8; ++r) {
        int m = m0 + mi * 16 + basem + r;
        out[(size_t)m * DIM + n] = acc[mi][t][r] + bv;
      }
    }
  }
}

// ---------------- launcher ----------------
extern "C" void kernel_launch(void* const* d_in, const int* in_sizes, int n_in,
                              void* d_out, int out_size, void* d_ws, size_t ws_size,
                              hipStream_t stream) {
  const float* x      = (const float*)d_in[0];   // [2,2048,1024]
  const float* w_qkv  = (const float*)d_in[1];   // [3072,1024]
  const float* w_proj = (const float*)d_in[2];   // [1024,1024]
  const float* b_proj = (const float*)d_in[3];   // [1024]
  float* out = (float*)d_out;

  char* ws = (char*)d_ws;
  __bf16* Xb  = (__bf16*)(ws);                        //  8 MiB [4096,1024]
  __bf16* Wqb = (__bf16*)(ws + (size_t) 8 * 1048576); //  6 MiB [3072,1024]
  __bf16* Wpb = (__bf16*)(ws + (size_t)14 * 1048576); //  2 MiB [1024,1024]
  __bf16* Qb  = (__bf16*)(ws + (size_t)16 * 1048576); //  8 MiB [32,2048,64]
  __bf16* Kb  = (__bf16*)(ws + (size_t)24 * 1048576); //  8 MiB [32,2048,64]
  __bf16* Vt  = (__bf16*)(ws + (size_t)32 * 1048576); //  8 MiB [32,64,2048]
  __bf16* Ob  = (__bf16*)(ws + (size_t)40 * 1048576); //  8 MiB [4096,1024]

  cvt_bf16<<<4096, 256, 0, stream>>>(x,      Xb,  (BATCH * LQ * DIM) / 4);
  cvt_bf16<<<3072, 256, 0, stream>>>(w_qkv,  Wqb, (3 * DIM * DIM) / 4);
  cvt_bf16<<<1024, 256, 0, stream>>>(w_proj, Wpb, (DIM * DIM) / 4);

  qkv_gemm<<<dim3(12, 64), 256, 0, stream>>>(Xb, Wqb, Qb, Kb, Vt);
  attn_kernel<<<dim3(LQ / 64, NH, BATCH), 128, 0, stream>>>(Qb, Kb, Vt, Ob);
  proj_gemm<<<dim3(4, 64), 256, 0, stream>>>(Ob, Wpb, b_proj, out);
}
